// MultiheadEinsumOrig_6485400617573
// MI455X (gfx1250) — matrix operand order repaired and checked by reading, not compiled, not run on hardware
//
#include <hip/hip_runtime.h>

// ---------------------------------------------------------------------------
// Types for CDNA5 WMMA (wave32, 16x16x32 bf16 -> f32)
// ---------------------------------------------------------------------------
typedef __bf16 bf16_t;
typedef __attribute__((ext_vector_type(16))) __bf16 v16bf;
typedef __attribute__((ext_vector_type(8)))  __bf16 v8bf;
typedef __attribute__((ext_vector_type(4)))  __bf16 v4bf;
typedef __attribute__((ext_vector_type(8)))  float  v8f;

#define WMMA_BF16(a, b, c) \
  __builtin_amdgcn_wmma_f32_16x16x32_bf16(false, (a), false, (b), (short)0, (c), false, false)

// Problem dimensions (match the reference)
constexpr int Bn = 4;
constexpr int Sn = 2048;
constexpr int En = 1024;
constexpr int Hn = 16;
constexpr int Dn = 64;        // head dim
constexpr int Fn = 3 * En;    // 3072 fused QKV features

// Workspace layout (bytes)
constexpr size_t OFF_XB  = 0;                                   // x bf16       [B*S, E]
constexpr size_t OFF_WQ  = OFF_XB  + (size_t)Bn*Sn*En*2;        // Wqkv bf16    [3E, E]
constexpr size_t OFF_WO  = OFF_WQ  + (size_t)Fn*En*2;           // Wout bf16    [E, E]
constexpr size_t OFF_Q   = OFF_WO  + (size_t)En*En*2;           // Q bf16       [B,H,S,D] (pre-scaled by D^-0.5)
constexpr size_t OFF_K   = OFF_Q   + (size_t)Bn*Hn*Sn*Dn*2;     // K bf16       [B,H,S,D]
constexpr size_t OFF_VT  = OFF_K   + (size_t)Bn*Hn*Sn*Dn*2;     // V^T bf16     [B,H,D,S]
constexpr size_t OFF_AO  = OFF_VT  + (size_t)Bn*Hn*Sn*Dn*2;     // attn out bf16 [B,S,E]

__device__ __forceinline__ v8f zero8() {
  v8f z = {0.f, 0.f, 0.f, 0.f, 0.f, 0.f, 0.f, 0.f};
  return z;
}

// ---------------------------------------------------------------------------
// A-operand loader: 16x32 bf16 tile, row-major source (row stride ld elems).
// ISA layout: lane L (r=L&15, kh=L>>4) holds row (row+r), K chunks
// [col+kh*8 .. +7] and [col+16+kh*8 .. +7]   -> two 16-byte loads.
// ---------------------------------------------------------------------------
__device__ __forceinline__ v16bf load_a_bf16(const bf16_t* base, int ld, int row, int col) {
  const int lane = threadIdx.x & 31;
  const int r  = lane & 15;
  const int kh = lane >> 4;
  const bf16_t* p = base + (size_t)(row + r) * ld + col + kh * 8;
  v8bf lo = *(const v8bf*)p;
  v8bf hi = *(const v8bf*)(p + 16);
  v16bf out;
#pragma unroll
  for (int i = 0; i < 8; ++i) { out[i] = lo[i]; out[i + 8] = hi[i]; }
  return out;
}

// ---------------------------------------------------------------------------
// B-operand loader: 32x16 bf16 tile where B[k][n] = src[row+n][col+k]
// (source row-major, row stride ld).  ISA layout: lane L (n=L&15, kh=L>>4)
// holds B[kh*8+i][n] and B[16+kh*8+i][n] for i=0..7 -> two 16-byte loads.
// ---------------------------------------------------------------------------
__device__ __forceinline__ v16bf load_b_bf16(const bf16_t* base, int ld, int row, int col) {
  const int lane = threadIdx.x & 31;
  const int n  = lane & 15;
  const int kh = lane >> 4;
  const bf16_t* p = base + (size_t)(row + n) * ld + col + kh * 8;
  v8bf lo = *(const v8bf*)p;
  v8bf hi = *(const v8bf*)(p + 16);
  v16bf out;
#pragma unroll
  for (int i = 0; i < 8; ++i) { out[i] = lo[i]; out[i + 8] = hi[i]; }
  return out;
}

// ---------------------------------------------------------------------------
// CDNA5 async global->LDS copy (16 bytes / lane), tracked by ASYNCcnt.
// lds_off is the LDS byte address (low 32 bits of a generic LDS pointer).
// ---------------------------------------------------------------------------
__device__ __forceinline__ void async_g2l_b128(unsigned lds_off, const bf16_t* g) {
  asm volatile("global_load_async_to_lds_b128 %0, %1, off"
               :: "v"(lds_off), "v"(g) : "memory");
}
__device__ __forceinline__ void wait_asynccnt0() {
  asm volatile("s_wait_asynccnt 0" ::: "memory");
}

// ---------------------------------------------------------------------------
// Kernel 0: fp32 -> bf16 conversion (4 elems / thread, packed 8-byte store)
// ---------------------------------------------------------------------------
__global__ void cvt_f32_bf16_kernel(const float* __restrict__ in, bf16_t* __restrict__ out, int n) {
  int i = (blockIdx.x * blockDim.x + threadIdx.x) * 4;
  if (i + 3 < n) {
    float4 v = *(const float4*)(in + i);
    v4bf o;
    o[0] = (bf16_t)v.x; o[1] = (bf16_t)v.y; o[2] = (bf16_t)v.z; o[3] = (bf16_t)v.w;
    *(v4bf*)(out + i) = o;
  }
}

// ---------------------------------------------------------------------------
// Kernel 1: fused QKV projection   proj[m,f] = sum_e x[m,e] * Wqkv[f,e]
// Wave tile: 32 (m) x 64 (f).  Epilogue decodes f = d*48 + t*16 + h and
// scatters to Q[B,H,S,D] (scaled by D^-0.5), K[B,H,S,D], V^T[B,H,D,S].
// ---------------------------------------------------------------------------
__global__ void __launch_bounds__(256)
qkv_gemm_kernel(const bf16_t* __restrict__ xb, const bf16_t* __restrict__ wq,
                bf16_t* __restrict__ Qd, bf16_t* __restrict__ Kd, bf16_t* __restrict__ Vt) {
  const int wave  = threadIdx.x >> 5;
  const int wt    = blockIdx.x * 8 + wave;
  const int FT    = Fn / 64;              // 48 f-tiles
  const int mbase = (wt / FT) * 32;
  const int fbase = (wt % FT) * 64;
  const int lane  = threadIdx.x & 31;
  const int kh    = lane >> 4;
  const int h     = lane & 15;

  v8f acc[2][4];
#pragma unroll
  for (int i = 0; i < 2; ++i)
#pragma unroll
    for (int j = 0; j < 4; ++j) acc[i][j] = zero8();

  for (int e0 = 0; e0 < En; e0 += 32) {
    v16bf a0 = load_a_bf16(xb, En, mbase,      e0);
    v16bf a1 = load_a_bf16(xb, En, mbase + 16, e0);
    v16bf b0 = load_b_bf16(wq, En, fbase,      e0);
    v16bf b1 = load_b_bf16(wq, En, fbase + 16, e0);
    v16bf b2 = load_b_bf16(wq, En, fbase + 32, e0);
    v16bf b3 = load_b_bf16(wq, En, fbase + 48, e0);
    acc[0][0] = WMMA_BF16(a0, b0, acc[0][0]);
    acc[0][1] = WMMA_BF16(a0, b1, acc[0][1]);
    acc[0][2] = WMMA_BF16(a0, b2, acc[0][2]);
    acc[0][3] = WMMA_BF16(a0, b3, acc[0][3]);
    acc[1][0] = WMMA_BF16(a1, b0, acc[1][0]);
    acc[1][1] = WMMA_BF16(a1, b1, acc[1][1]);
    acc[1][2] = WMMA_BF16(a1, b2, acc[1][2]);
    acc[1][3] = WMMA_BF16(a1, b3, acc[1][3]);
  }

#pragma unroll
  for (int mt = 0; mt < 2; ++mt) {
#pragma unroll
    for (int nt = 0; nt < 4; ++nt) {
      const int ft = (fbase + nt * 16) >> 4;  // 16-wide f-tile index
      const int d  = ft / 3;                  // head-dim index
      const int t  = ft % 3;                  // 0=q 1=k 2=v
      const float scale = (t == 0) ? 0.125f : 1.0f;   // fold D^-0.5 into q
#pragma unroll
      for (int r = 0; r < 8; ++r) {
        const int m    = mbase + mt * 16 + r + 8 * kh;
        const int bidx = m >> 11;            // / S
        const int s    = m & (Sn - 1);
        const bf16_t val = (bf16_t)(acc[mt][nt][r] * scale);
        if (t == 0)
          Qd[((size_t)(bidx * Hn + h) * Sn + s) * Dn + d] = val;
        else if (t == 1)
          Kd[((size_t)(bidx * Hn + h) * Sn + s) * Dn + d] = val;
        else
          Vt[((size_t)(bidx * Hn + h) * Dn + d) * Sn + s] = val;
      }
    }
  }
}

// ---------------------------------------------------------------------------
// Kernel 2: flash attention.  4 waves/block share one (b,h); per 32-key step
// the block stages K (32x64) and V^T (64x32) tiles into double-buffered LDS
// with global_load_async_to_lds_b128 (ASYNCcnt), overlapped with compute.
// Each wave: 4 WMMAs for scores (K=64), online softmax, P staged through
// wave-private LDS (C-layout -> A-layout), 4 WMMAs for P.V^T.
// ---------------------------------------------------------------------------
__global__ void __launch_bounds__(128)
attn_kernel(const bf16_t* __restrict__ Qd, const bf16_t* __restrict__ Kd,
            const bf16_t* __restrict__ Vt, bf16_t* __restrict__ AO) {
  __shared__ bf16_t kbuf[2][32 * 64];   // keys x e        (4KB each)
  __shared__ bf16_t vbuf[2][64 * 32];   // d    x keys     (4KB each)
  __shared__ bf16_t pstage[4][16 * 32]; // per-wave P tile (4KB total)

  const int tid   = threadIdx.x;        // 0..127
  const int wave  = tid >> 5;
  const int lane  = tid & 31;
  const int kh    = lane >> 4;
  const int nlo   = lane & 15;
  const int bh    = blockIdx.y;
  const int b     = bh / Hn;
  const int h     = bh % Hn;
  const int qbase = (blockIdx.x * 4 + wave) * 16;

  const bf16_t* Qp = Qd + (size_t)(b * Hn + h) * Sn * Dn;
  const bf16_t* Kp = Kd + (size_t)(b * Hn + h) * Sn * Dn;
  const bf16_t* Vp = Vt + (size_t)(b * Hn + h) * Dn * Sn;

  // Cooperative async stage of one 32-key step: 256+256 16B chunks, 128 thr.
  auto issue_tiles = [&](int j, int p) {
#pragma unroll
    for (int c0 = 0; c0 < 2; ++c0) {
      const int c = tid + c0 * 128;                 // K tile chunk: 32 rows x 128B
      const bf16_t* g = Kp + (size_t)(j + (c >> 3)) * Dn + (c & 7) * 8;
      async_g2l_b128((unsigned)(size_t)(&kbuf[p][0]) + c * 16, g);
    }
#pragma unroll
    for (int c0 = 0; c0 < 2; ++c0) {
      const int c = tid + c0 * 128;                 // V^T tile chunk: 64 rows x 64B
      const bf16_t* g = Vp + (size_t)(c >> 2) * Sn + j + (c & 3) * 8;
      async_g2l_b128((unsigned)(size_t)(&vbuf[p][0]) + c * 16, g);
    }
  };

  const v16bf a0 = load_a_bf16(Qp, Dn, qbase, 0);    // e = 0..31  (q pre-scaled)
  const v16bf a1 = load_a_bf16(Qp, Dn, qbase, 32);   // e = 32..63

  v8f o[4];
#pragma unroll
  for (int nt = 0; nt < 4; ++nt) o[nt] = zero8();
  float mrow[8], lrow[8];
#pragma unroll
  for (int r = 0; r < 8; ++r) { mrow[r] = -3.0e38f; lrow[r] = 0.f; }

  issue_tiles(0, 0);

  for (int j = 0, it = 0; j < Sn; j += 32, ++it) {
    const int p = it & 1;
    wait_asynccnt0();       // my async writes for buffer p are done
    __syncthreads();        // everyone's are; everyone finished reading p^1
    if (j + 32 < Sn) issue_tiles(j + 32, p ^ 1);   // prefetch next step

    const bf16_t* kb = &kbuf[p][0];
    const bf16_t* vb = &vbuf[p][0];

    // ---- scores: two 16x16 tiles over keys j..j+31, K-dim = D = 64 ----
    v16bf kb00 = load_b_bf16(kb, Dn, 0,  0);
    v16bf kb01 = load_b_bf16(kb, Dn, 0,  32);
    v16bf kb10 = load_b_bf16(kb, Dn, 16, 0);
    v16bf kb11 = load_b_bf16(kb, Dn, 16, 32);
    v8f s0 = WMMA_BF16(a0, kb00, zero8());
    s0     = WMMA_BF16(a1, kb01, s0);
    v8f s1 = WMMA_BF16(a0, kb10, zero8());
    s1     = WMMA_BF16(a1, kb11, s1);

    // ---- online softmax update (row = r + 8*kh, data across lanes 0..15) --
#pragma unroll
    for (int r = 0; r < 8; ++r) {
      float tmax = fmaxf(s0[r], s1[r]);
#pragma unroll
      for (int msk = 8; msk >= 1; msk >>= 1)
        tmax = fmaxf(tmax, __shfl_xor(tmax, msk, 32));
      const float mnew = fmaxf(mrow[r], tmax);
      const float corr = __expf(mrow[r] - mnew);
      mrow[r] = mnew;
      const float p0 = __expf(s0[r] - mnew);
      const float p1 = __expf(s1[r] - mnew);
      float rs = p0 + p1;
#pragma unroll
      for (int msk = 8; msk >= 1; msk >>= 1)
        rs += __shfl_xor(rs, msk, 32);
      lrow[r] = lrow[r] * corr + rs;
#pragma unroll
      for (int nt = 0; nt < 4; ++nt) o[nt][r] *= corr;
      pstage[wave][(r + 8 * kh) * 32 + nlo]      = (bf16_t)p0;
      pstage[wave][(r + 8 * kh) * 32 + 16 + nlo] = (bf16_t)p1;
    }

    // wave-private LDS re-swizzle: DS ops are in-order per wave; keep the
    // compiler from reordering and make the dependency explicit.
    asm volatile("s_wait_dscnt 0" ::: "memory");

    const v16bf pa = load_a_bf16(&pstage[wave][0], 32, 0, 0);  // P as A operand
#pragma unroll
    for (int nt = 0; nt < 4; ++nt) {
      v16bf bv = load_b_bf16(vb, 32, nt * 16, 0);              // V^T rows = d
      o[nt] = WMMA_BF16(pa, bv, o[nt]);
    }
  }

  // ---- normalize and store bf16 to AO[b, s, h*D + d] ----
#pragma unroll
  for (int r = 0; r < 8; ++r) {
    const float inv = 1.0f / lrow[r];
    const int s = qbase + r + 8 * kh;
#pragma unroll
    for (int nt = 0; nt < 4; ++nt) {
      AO[((size_t)b * Sn + s) * En + h * Dn + nt * 16 + nlo] = (bf16_t)(o[nt][r] * inv);
    }
  }
}

// ---------------------------------------------------------------------------
// Kernel 3: output projection  out[m,f] = sum_e AO[m,e] * Wout[f,e]  (f32 out)
// ---------------------------------------------------------------------------
__global__ void __launch_bounds__(256)
outproj_kernel(const bf16_t* __restrict__ AO, const bf16_t* __restrict__ Wo,
               float* __restrict__ out) {
  const int wave  = threadIdx.x >> 5;
  const int wt    = blockIdx.x * 8 + wave;
  const int NT    = En / 64;              // 16 f-tiles
  const int mbase = (wt / NT) * 32;
  const int nbase = (wt % NT) * 64;
  const int lane  = threadIdx.x & 31;
  const int kh    = lane >> 4;
  const int nlo   = lane & 15;

  v8f acc[2][4];
#pragma unroll
  for (int i = 0; i < 2; ++i)
#pragma unroll
    for (int j = 0; j < 4; ++j) acc[i][j] = zero8();

  for (int e0 = 0; e0 < En; e0 += 32) {
    v16bf a0 = load_a_bf16(AO, En, mbase,      e0);
    v16bf a1 = load_a_bf16(AO, En, mbase + 16, e0);
    v16bf b0 = load_b_bf16(Wo, En, nbase,      e0);
    v16bf b1 = load_b_bf16(Wo, En, nbase + 16, e0);
    v16bf b2 = load_b_bf16(Wo, En, nbase + 32, e0);
    v16bf b3 = load_b_bf16(Wo, En, nbase + 48, e0);
    acc[0][0] = WMMA_BF16(a0, b0, acc[0][0]);
    acc[0][1] = WMMA_BF16(a0, b1, acc[0][1]);
    acc[0][2] = WMMA_BF16(a0, b2, acc[0][2]);
    acc[0][3] = WMMA_BF16(a0, b3, acc[0][3]);
    acc[1][0] = WMMA_BF16(a1, b0, acc[1][0]);
    acc[1][1] = WMMA_BF16(a1, b1, acc[1][1]);
    acc[1][2] = WMMA_BF16(a1, b2, acc[1][2]);
    acc[1][3] = WMMA_BF16(a1, b3, acc[1][3]);
  }

#pragma unroll
  for (int mt = 0; mt < 2; ++mt)
#pragma unroll
    for (int nt = 0; nt < 4; ++nt)
#pragma unroll
      for (int r = 0; r < 8; ++r) {
        const int m = mbase + mt * 16 + r + 8 * kh;
        out[(size_t)m * En + nbase + nt * 16 + nlo] = acc[mt][nt][r];
      }
}

// ---------------------------------------------------------------------------
// Host launcher
// ---------------------------------------------------------------------------
extern "C" void kernel_launch(void* const* d_in, const int* in_sizes, int n_in,
                              void* d_out, int out_size, void* d_ws, size_t ws_size,
                              hipStream_t stream) {
  const float* x    = (const float*)d_in[0];
  const float* Wqkv = (const float*)d_in[1];
  const float* Wout = (const float*)d_in[2];
  float* out        = (float*)d_out;
  char*  ws         = (char*)d_ws;

  bf16_t* xb  = (bf16_t*)(ws + OFF_XB);
  bf16_t* wqb = (bf16_t*)(ws + OFF_WQ);
  bf16_t* wob = (bf16_t*)(ws + OFF_WO);
  bf16_t* Qw  = (bf16_t*)(ws + OFF_Q);
  bf16_t* Kw  = (bf16_t*)(ws + OFF_K);
  bf16_t* Vtw = (bf16_t*)(ws + OFF_VT);
  bf16_t* AOw = (bf16_t*)(ws + OFF_AO);

  const int nx  = Bn * Sn * En;   // 8,388,608
  const int nwq = Fn * En;        // 3,145,728
  const int nwo = En * En;        // 1,048,576

  cvt_f32_bf16_kernel<<<nx  / (256 * 4), 256, 0, stream>>>(x,    xb,  nx);
  cvt_f32_bf16_kernel<<<nwq / (256 * 4), 256, 0, stream>>>(Wqkv, wqb, nwq);
  cvt_f32_bf16_kernel<<<nwo / (256 * 4), 256, 0, stream>>>(Wout, wob, nwo);

  // QKV GEMM: (B*S/32) * (3E/64) = 256*48 = 12288 wave tiles / 8 waves
  qkv_gemm_kernel<<<1536, 256, 0, stream>>>(xb, wqb, Qw, Kw, Vtw);

  // Attention: grid (S/64 query chunks, B*H heads), 4 waves/block
  attn_kernel<<<dim3(Sn / 64, Bn * Hn), 128, 0, stream>>>(Qw, Kw, Vtw, AOw);

  // Output GEMM: 256 * 16 = 4096 wave tiles / 8 waves
  outproj_kernel<<<512, 256, 0, stream>>>(AOw, wob, out);
}
